// WindowAttention_3813930959453
// MI455X (gfx1250) — matrix operand order repaired and checked by reading, not compiled, not run on hardware
//
#include <hip/hip_runtime.h>
#include <hip/hip_bf16.h>

typedef __bf16 bf16_t;
typedef __attribute__((ext_vector_type(16))) __bf16 v16bf;
typedef __attribute__((ext_vector_type(8)))  __bf16 v8bf;
typedef __attribute__((ext_vector_type(4)))  __bf16 v4bf;
typedef __attribute__((ext_vector_type(8)))  float  v8f;

#define DIMC     192
#define NHEADS   6
#define HDIM     32
#define NTOK     343
#define NPAD     352          // 22 tiles of 16
#define BWIN     256
#define QKV_COLS 576
#define ROWS     (BWIN * NTOK)     // 87808 = 16 * 5488
#define MTILES   (ROWS / 16)       // 5488
#define QSCALE   0.17677669529663687f   // 32^-0.5

// Async-to-LDS path (CDNA5 GLOBAL_LOAD_ASYNC_TO_LDS_B128 + s_wait_asynccnt),
// guarded so the file compiles on toolchains that do not declare the builtins.
#if defined(__HIP_DEVICE_COMPILE__)
#if __has_builtin(__builtin_amdgcn_global_load_async_to_lds_b128) && \
    __has_builtin(__builtin_amdgcn_s_wait_asynccnt)
#define HAVE_ASYNC_LDS 1
#endif
#endif

#ifdef HAVE_ASYNC_LDS
// Builtin signature (from compiler diagnostic): parameters are pointers to
// 16-byte int vectors in addrspace(1) (global) and addrspace(3) (LDS).
typedef __attribute__((__vector_size__(4 * sizeof(int)))) int i32x4;
typedef __attribute__((address_space(1))) i32x4* gbl_i32x4_p;
typedef __attribute__((address_space(3))) i32x4* lds_i32x4_p;
#endif

// ---------------- workspace layout ----------------
__host__ __device__ constexpr size_t align256(size_t x) { return (x + 255) & ~size_t(255); }
constexpr size_t SZ_XB    = align256((size_t)ROWS * DIMC * 2);                 // x in bf16
constexpr size_t SZ_WQKV  = align256((size_t)QKV_COLS * DIMC * 2);             // qkv_w bf16 [576][192]
constexpr size_t SZ_WPROJ = align256((size_t)DIMC * DIMC * 2);                 // proj_w bf16 [192][192]
constexpr size_t SZ_QKVB  = align256((size_t)BWIN * NHEADS * NPAD * HDIM * 2); // per q/k/v cache
constexpr size_t SZ_BIAS  = align256((size_t)NHEADS * NTOK * NTOK * 4);        // bias matrix f32
constexpr size_t SZ_ATTN  = align256((size_t)ROWS * DIMC * 2);                 // attn output bf16

constexpr size_t OFF_XB    = 0;
constexpr size_t OFF_WQKV  = OFF_XB    + SZ_XB;
constexpr size_t OFF_WPROJ = OFF_WQKV  + SZ_WQKV;
constexpr size_t OFF_Q     = OFF_WPROJ + SZ_WPROJ;
constexpr size_t OFF_K     = OFF_Q     + SZ_QKVB;
constexpr size_t OFF_VT    = OFF_K     + SZ_QKVB;
constexpr size_t OFF_BIAS  = OFF_VT    + SZ_QKVB;
constexpr size_t OFF_ATTN  = OFF_BIAS  + SZ_BIAS;

// ---------------- WMMA fragment loaders ----------------
// A fragment, 16x32 bf16 (M x K), src row-major [M][>=32] at (row0,k0) pre-offset.
// Lanes 0-15: M=lane, K={0..7, 16..23}; lanes 16-31: M=lane-16, K={8..15, 24..31}.
__device__ __forceinline__ v16bf load_a_frag(const bf16_t* p, int ld, int half, int lr) {
  const bf16_t* r = p + lr * ld + half * 8;
  v8bf lo = *reinterpret_cast<const v8bf*>(r);
  v8bf hi = *reinterpret_cast<const v8bf*>(r + 16);
  v16bf out;
#pragma unroll
  for (int i = 0; i < 8; ++i) { out[i] = lo[i]; out[i + 8] = hi[i]; }
  return out;
}

// B fragment, 32x16 bf16 (K x N), source stored as [N][K] row-major.
// Lanes 0-15: column N=lane, K={0..7, 16..23}; lanes 16-31: column N=lane-16, K={8..15, 24..31}.
__device__ __forceinline__ v16bf load_b_frag(const bf16_t* p, int ld, int half, int lr) {
  const bf16_t* r = p + lr * ld + half * 8;
  v8bf lo = *reinterpret_cast<const v8bf*>(r);
  v8bf hi = *reinterpret_cast<const v8bf*>(r + 16);
  v16bf out;
#pragma unroll
  for (int i = 0; i < 8; ++i) { out[i] = lo[i]; out[i + 8] = hi[i]; }
  return out;
}

__device__ __forceinline__ v8f wmma_bf16(v16bf a, v16bf b, v8f c) {
  return __builtin_amdgcn_wmma_f32_16x16x32_bf16(false, a, false, b, (short)0, c,
                                                 false, false);
}

#ifdef HAVE_ASYNC_LDS
// Copy one 16x32 bf16 K-tile (1024 B) global -> LDS with 2 async b128 ops/lane.
__device__ __forceinline__ void async_copy_ktile(const bf16_t* gsrc, bf16_t* ldst,
                                                 int kt, int lane) {
  const char* g = (const char*)(gsrc + kt * 16 * HDIM) + lane * 16;
  char* l = (char*)ldst + lane * 16;
  __builtin_amdgcn_global_load_async_to_lds_b128(
      (gbl_i32x4_p)g, (lds_i32x4_p)l, 0, 0);
  __builtin_amdgcn_global_load_async_to_lds_b128(
      (gbl_i32x4_p)(g + 512), (lds_i32x4_p)(l + 512), 0, 0);
}
#endif

// ---------------- prep kernels ----------------
__global__ void k_convert(const float* __restrict__ x, const float* __restrict__ wq,
                          const float* __restrict__ wp, bf16_t* __restrict__ xb,
                          bf16_t* __restrict__ wqb, bf16_t* __restrict__ wpb) {
  long i = (long)blockIdx.x * 256 + threadIdx.x;
  if (i < (long)ROWS * DIMC) xb[i] = (bf16_t)x[i];
  if (i < QKV_COLS * DIMC)   wqb[i] = (bf16_t)wq[i];
  if (i < DIMC * DIMC)       wpb[i] = (bf16_t)wp[i];
}

__global__ void k_bias(const float* __restrict__ table, float* __restrict__ biasmat) {
  int t = blockIdx.x * 256 + threadIdx.x;
  if (t >= NTOK * NTOK) return;
  int q = t / NTOK, k = t - q * NTOK;
  int qd = q / 49, qr = q - qd * 49, qh = qr / 7, qw = qr - qh * 7;
  int kd = k / 49, kr = k - kd * 49, kh = kr / 7, kw = kr - kh * 7;
  int idx = (qd - kd + 6) * 169 + (qh - kh + 6) * 13 + (qw - kw + 6);
#pragma unroll
  for (int h = 0; h < NHEADS; ++h)
    biasmat[((long)h * NTOK + q) * NTOK + k] = table[idx * NHEADS + h];
}

__global__ void k_zeropad(bf16_t* __restrict__ qb, bf16_t* __restrict__ kb,
                          bf16_t* __restrict__ vt) {
  long bh = blockIdx.x;              // window*head, 1536 total
  int  t  = threadIdx.x;
  bf16_t z = (bf16_t)0.0f;
  for (int i = t; i < 9 * HDIM; i += 256) {   // q/k pad rows 343..351
    int r = NTOK + i / HDIM, c = i % HDIM;
    qb[(bh * NPAD + r) * HDIM + c] = z;
    kb[(bh * NPAD + r) * HDIM + c] = z;
  }
  for (int i = t; i < HDIM * 9; i += 256) {   // v^T pad keys 343..351
    int d = i / 9, kk = NTOK + i % 9;
    vt[(bh * HDIM + d) * NPAD + kk] = z;
  }
}

// ---------------- QKV projection GEMM: 16x64 per wave ----------------
__device__ __forceinline__ void scatter_qkv(v8f acc, int mt, int col, int half,
                                            const float* __restrict__ qkv_b,
                                            bf16_t* __restrict__ qb,
                                            bf16_t* __restrict__ kb,
                                            bf16_t* __restrict__ vt) {
  int which = col / DIMC, d = col - which * DIMC;
  int head = d >> 5, hd = d & 31;
  float bias = qkv_b[col];
#pragma unroll
  for (int j = 0; j < 8; ++j) {
    int m = mt * 16 + j + 8 * half;
    int bwin = m / NTOK, n = m - bwin * NTOK;
    float v = acc[j] + bias;
    long bh = (long)bwin * NHEADS + head;
    if (which == 0)      qb[(bh * NPAD + n) * HDIM + hd] = (bf16_t)(v * QSCALE);
    else if (which == 1) kb[(bh * NPAD + n) * HDIM + hd] = (bf16_t)v;
    else                 vt[(bh * HDIM + hd) * NPAD + n] = (bf16_t)v;
  }
}

__global__ __launch_bounds__(32) void k_qkv(const bf16_t* __restrict__ xb,
                                            const bf16_t* __restrict__ wqb,
                                            const float* __restrict__ qkv_b,
                                            bf16_t* __restrict__ qb,
                                            bf16_t* __restrict__ kb,
                                            bf16_t* __restrict__ vt) {
  int mt = blockIdx.x, ng = blockIdx.y;               // ng: 9 groups of 64 cols
  int lane = threadIdx.x, half = lane >> 4, lr = lane & 15;
  const bf16_t* arow = xb  + (long)(mt * 16) * DIMC;
  const bf16_t* brow = wqb + (long)(ng * 64) * DIMC;
  v8f acc0 = {}, acc1 = {}, acc2 = {}, acc3 = {};
#pragma unroll
  for (int kk = 0; kk < 6; ++kk) {
    if (kk < 5) __builtin_prefetch(brow + (kk + 1) * 32 + lr * DIMC, 0, 0);
    v16bf a  = load_a_frag(arow + kk * 32, DIMC, half, lr);
    v16bf b0 = load_b_frag(brow + kk * 32, DIMC, half, lr);
    v16bf b1 = load_b_frag(brow + 16 * DIMC + kk * 32, DIMC, half, lr);
    v16bf b2 = load_b_frag(brow + 32 * DIMC + kk * 32, DIMC, half, lr);
    v16bf b3 = load_b_frag(brow + 48 * DIMC + kk * 32, DIMC, half, lr);
    acc0 = wmma_bf16(a, b0, acc0);
    acc1 = wmma_bf16(a, b1, acc1);
    acc2 = wmma_bf16(a, b2, acc2);
    acc3 = wmma_bf16(a, b3, acc3);
  }
  scatter_qkv(acc0, mt, ng * 64 + lr,      half, qkv_b, qb, kb, vt);
  scatter_qkv(acc1, mt, ng * 64 + 16 + lr, half, qkv_b, qb, kb, vt);
  scatter_qkv(acc2, mt, ng * 64 + 32 + lr, half, qkv_b, qb, kb, vt);
  scatter_qkv(acc3, mt, ng * 64 + 48 + lr, half, qkv_b, qb, kb, vt);
}

// ---------------- attention core: one wave per (qtile, head, window) ----------
__global__ __launch_bounds__(32) void k_attn(const bf16_t* __restrict__ qb,
                                             const bf16_t* __restrict__ kb,
                                             const bf16_t* __restrict__ vt,
                                             const float* __restrict__ biasmat,
                                             const float* __restrict__ mask,
                                             bf16_t* __restrict__ attnout) {
  __shared__ float  s_s[16 * NPAD];   // f32 scores, 16 x 352
  __shared__ bf16_t s_p[16 * NPAD];   // bf16 softmax numerators
#ifdef HAVE_ASYNC_LDS
  __shared__ bf16_t s_kt[2][16 * HDIM];  // double-buffered K tiles
#endif
  int qt = blockIdx.x, h = blockIdx.y, bw = blockIdx.z;
  int lane = threadIdx.x, half = lane >> 4, lr = lane & 15;
  long bh = (long)bw * NHEADS + h;

  v16bf qa = load_a_frag(qb + (bh * NPAD + qt * 16) * HDIM, HDIM, half, lr);
  const float* brow = biasmat + (long)h * NTOK * NTOK;
  const float* mrow = mask + (long)(bw & 63) * NTOK * NTOK;
  const bf16_t* kpanel = kb + bh * NPAD * HDIM;

#ifdef HAVE_ASYNC_LDS
  async_copy_ktile(kpanel, s_kt[0], 0, lane);   // prologue: stage tile 0
#endif

  // pass 1: scores = (q*scale) . k^T + bias + mask  -> LDS
  for (int kt = 0; kt < NPAD / 16; ++kt) {
#ifdef HAVE_ASYNC_LDS
    int cur = kt & 1;
    if (kt + 1 < NPAD / 16) {
      async_copy_ktile(kpanel, s_kt[cur ^ 1], kt + 1, lane);
      __builtin_amdgcn_s_wait_asynccnt(2);   // tile kt's 2 ops complete
    } else {
      __builtin_amdgcn_s_wait_asynccnt(0);
    }
    __builtin_amdgcn_wave_barrier();
    v16bf kf = load_b_frag(s_kt[cur], HDIM, half, lr);
#else
    if (kt + 1 < NPAD / 16)
      __builtin_prefetch(kpanel + ((kt + 1) * 16 + lr) * HDIM, 0, 0);
    v16bf kf = load_b_frag(kpanel + kt * 16 * HDIM, HDIM, half, lr);
#endif
    v8f s = {};
    s = wmma_bf16(qa, kf, s);
#pragma unroll
    for (int j = 0; j < 8; ++j) {
      int qrow = qt * 16 + j + 8 * half;
      int kcol = kt * 16 + lr;
      float val = -1e30f;
      if (qrow < NTOK && kcol < NTOK)
        val = s[j] + brow[(long)qrow * NTOK + kcol] + mrow[(long)qrow * NTOK + kcol];
      s_s[(j + 8 * half) * NPAD + kcol] = val;
    }
  }
  __syncthreads();

  // pass 2: per-row max & sum (half-wave owns half the columns of each row),
  // vectorized float4 LDS reads + packed 4xbf16 stores of the numerators.
  int row = lane & 15;
  int c0  = (lane >> 4) * (NPAD / 2);
  const float4* srow = reinterpret_cast<const float4*>(&s_s[row * NPAD + c0]);
  float mx = -1e30f;
  for (int c = 0; c < NPAD / 8; ++c) {
    float4 v = srow[c];
    mx = fmaxf(mx, fmaxf(fmaxf(v.x, v.y), fmaxf(v.z, v.w)));
  }
  mx = fmaxf(mx, __shfl_xor(mx, 16, 32));
  float sum = 0.0f;
  for (int c = 0; c < NPAD / 8; ++c) {
    float4 v = srow[c];
    float p0 = __expf(v.x - mx), p1 = __expf(v.y - mx);
    float p2 = __expf(v.z - mx), p3 = __expf(v.w - mx);
    v4bf pv = {(bf16_t)p0, (bf16_t)p1, (bf16_t)p2, (bf16_t)p3};
    *reinterpret_cast<v4bf*>(&s_p[row * NPAD + c0 + c * 4]) = pv;
    sum += (p0 + p1) + (p2 + p3);
  }
  sum += __shfl_xor(sum, 16, 32);
  float inv = 1.0f / sum;
  __syncthreads();

  // pass 3: O = P @ V  (pairs of key tiles -> K=32 per WMMA)
  v8f o0 = {}, o1 = {};
  const bf16_t* vb = vt + bh * HDIM * NPAD;
  for (int kp = 0; kp < NPAD / 32; ++kp) {
    v16bf pa = load_a_frag(s_p + kp * 32, NPAD, half, lr);
    v16bf b0 = load_b_frag(vb + kp * 32, NPAD, half, lr);              // hd 0..15
    v16bf b1 = load_b_frag(vb + 16 * NPAD + kp * 32, NPAD, half, lr);  // hd 16..31
    o0 = wmma_bf16(pa, b0, o0);
    o1 = wmma_bf16(pa, b1, o1);
  }
#pragma unroll
  for (int j = 0; j < 8; ++j) {
    int ml = j + 8 * half;
    int qrow = qt * 16 + ml;
    float invj = __shfl(inv, ml, 32);
    if (qrow < NTOK) {
      long r = (long)bw * NTOK + qrow;
      attnout[r * DIMC + h * HDIM + lr]      = (bf16_t)(o0[j] * invj);
      attnout[r * DIMC + h * HDIM + 16 + lr] = (bf16_t)(o1[j] * invj);
    }
  }
}

// ---------------- output projection GEMM: 16x64 per wave ----------------
__global__ __launch_bounds__(32) void k_proj(const bf16_t* __restrict__ attnout,
                                             const bf16_t* __restrict__ wpb,
                                             const float* __restrict__ proj_b,
                                             float* __restrict__ out) {
  int mt = blockIdx.x, ng = blockIdx.y;               // ng: 3 groups of 64 cols
  int lane = threadIdx.x, half = lane >> 4, lr = lane & 15;
  const bf16_t* arow = attnout + (long)(mt * 16) * DIMC;
  const bf16_t* brow = wpb + (long)(ng * 64) * DIMC;
  v8f acc0 = {}, acc1 = {}, acc2 = {}, acc3 = {};
#pragma unroll
  for (int kk = 0; kk < 6; ++kk) {
    if (kk < 5) __builtin_prefetch(arow + (kk + 1) * 32 + lr * DIMC, 0, 0);
    v16bf a  = load_a_frag(arow + kk * 32, DIMC, half, lr);
    v16bf b0 = load_b_frag(brow + kk * 32, DIMC, half, lr);
    v16bf b1 = load_b_frag(brow + 16 * DIMC + kk * 32, DIMC, half, lr);
    v16bf b2 = load_b_frag(brow + 32 * DIMC + kk * 32, DIMC, half, lr);
    v16bf b3 = load_b_frag(brow + 48 * DIMC + kk * 32, DIMC, half, lr);
    acc0 = wmma_bf16(a, b0, acc0);
    acc1 = wmma_bf16(a, b1, acc1);
    acc2 = wmma_bf16(a, b2, acc2);
    acc3 = wmma_bf16(a, b3, acc3);
  }
#pragma unroll
  for (int sub = 0; sub < 4; ++sub) {
    v8f acc = sub == 0 ? acc0 : sub == 1 ? acc1 : sub == 2 ? acc2 : acc3;
    int col = ng * 64 + sub * 16 + lr;
    float pb = proj_b[col];
#pragma unroll
    for (int j = 0; j < 8; ++j) {
      long m = (long)mt * 16 + j + 8 * half;
      out[m * DIMC + col] = acc[j] + pb;
    }
  }
}

// ---------------- launcher ----------------
extern "C" void kernel_launch(void* const* d_in, const int* in_sizes, int n_in,
                              void* d_out, int out_size, void* d_ws, size_t ws_size,
                              hipStream_t stream) {
  const float* x      = (const float*)d_in[0];
  const float* mask   = (const float*)d_in[1];
  const float* qkv_w  = (const float*)d_in[2];
  const float* qkv_b  = (const float*)d_in[3];
  const float* rpb    = (const float*)d_in[4];
  const float* proj_w = (const float*)d_in[5];
  const float* proj_b = (const float*)d_in[6];
  float* out = (float*)d_out;

  char* ws = (char*)d_ws;
  bf16_t* xb      = (bf16_t*)(ws + OFF_XB);
  bf16_t* wqb     = (bf16_t*)(ws + OFF_WQKV);
  bf16_t* wpb     = (bf16_t*)(ws + OFF_WPROJ);
  bf16_t* qb      = (bf16_t*)(ws + OFF_Q);
  bf16_t* kb      = (bf16_t*)(ws + OFF_K);
  bf16_t* vt      = (bf16_t*)(ws + OFF_VT);
  float*  biasmat = (float*)(ws + OFF_BIAS);
  bf16_t* attnout = (bf16_t*)(ws + OFF_ATTN);

  k_convert<<<((long)ROWS * DIMC + 255) / 256, 256, 0, stream>>>(x, qkv_w, proj_w,
                                                                 xb, wqb, wpb);
  k_bias<<<(NTOK * NTOK + 255) / 256, 256, 0, stream>>>(rpb, biasmat);
  k_zeropad<<<BWIN * NHEADS, 256, 0, stream>>>(qb, kb, vt);
  k_qkv<<<dim3(MTILES, QKV_COLS / 64), 32, 0, stream>>>(xb, wqb, qkv_b, qb, kb, vt);
  k_attn<<<dim3(NPAD / 16, NHEADS, BWIN), 32, 0, stream>>>(qb, kb, vt, biasmat,
                                                           mask, attnout);
  k_proj<<<dim3(MTILES, DIMC / 64), 32, 0, stream>>>(attnout, wpb, proj_b, out);
}
